// MultiModalSpikeFusion_14929306321751
// MI455X (gfx1250) — matrix operand order repaired and checked by reading, not compiled, not run on hardware
//
#include <hip/hip_runtime.h>
#include <stddef.h>
#include <stdint.h>

// ---------------------------------------------------------------------------
// Types
// ---------------------------------------------------------------------------
typedef unsigned short ushort_t;
typedef __attribute__((ext_vector_type(16))) __bf16 bf16x16;
typedef __attribute__((ext_vector_type(8)))  float  f32x8;
typedef __attribute__((ext_vector_type(4)))  unsigned int u32x4;
typedef __attribute__((ext_vector_type(4)))  float  f32x4;

struct Pack32 { u32x4 lo, hi; };   // 32 bytes == one bf16x16 WMMA fragment

static __device__ __forceinline__ bf16x16 make_frag(u32x4 lo, u32x4 hi) {
  Pack32 p; p.lo = lo; p.hi = hi;
  return __builtin_bit_cast(bf16x16, p);
}

static __device__ __forceinline__ ushort_t f2bf(float f) {
  unsigned u = __float_as_uint(f);
  unsigned r = u + 0x7fffu + ((u >> 16) & 1u);   // round-to-nearest-even
  return (ushort_t)(r >> 16);
}

// CDNA5 async global->LDS copy (16B per lane), tracked by ASYNCcnt.
// VDST VGPR carries the LDS byte address (low 32 bits of the generic address).
static __device__ __forceinline__ void async_ld16(void* lds, const void* gptr) {
  unsigned l = (unsigned)(uintptr_t)lds;
  unsigned long long g = (unsigned long long)(uintptr_t)gptr;
  asm volatile("global_load_async_to_lds_b128 %0, %1, off"
               :: "v"(l), "v"(g) : "memory");
}
static __device__ __forceinline__ void wait_async0() {
  asm volatile("s_wait_asynccnt 0" ::: "memory");
}

// Problem constants
#define BB 2
#define TT 32
#define SS 256
#define DD 512
#define HH 8
#define HDIM 64
#define NTOK (BB*TT*SS)            // 16384 rows
#define NTEL ((size_t)NTOK * DD)   // 8,388,608 elements per [B,T,S,D] tensor

// ---------------------------------------------------------------------------
// f32 -> bf16 conversion
// ---------------------------------------------------------------------------
__global__ void cvt_f32_bf16(const float* __restrict__ in, ushort_t* __restrict__ out, int n) {
  int i = blockIdx.x * blockDim.x + threadIdx.x;
  int stride = gridDim.x * blockDim.x;
  for (; i < n; i += stride) out[i] = f2bf(in[i]);
}

// ---------------------------------------------------------------------------
// bf16 GEMM: out[M,N] = alpha*(A[M,K] @ Bw[K,N] + bias) + beta*resid
// 128x128 block tile, 8 waves (4x2), wave tile 32x64, WMMA f32_16x16x32_bf16.
// Double-buffered LDS tiles staged with async global->LDS copies so the next
// K-tile streams in while the current one feeds the matrix pipes.
// ---------------------------------------------------------------------------
#define APITCH 40   // 32 + 8 pad (ushorts) -> 80B row, 16B aligned, bank-spread
#define BPITCH 136  // 128 + 8 pad         -> 272B row, 16B aligned

__global__ __launch_bounds__(256) void gemm_bf16(
    const ushort_t* __restrict__ A, const ushort_t* __restrict__ Bw,
    const float* __restrict__ bias, const float* __restrict__ resid,
    const float* __restrict__ alpha_p, const float* __restrict__ beta_p,
    float* __restrict__ outf, ushort_t* __restrict__ outb,
    int M, int N, int K)
{
  __shared__ ushort_t As[2][128 * APITCH];
  __shared__ ushort_t Bs[2][32 * BPITCH];

  const int tid  = threadIdx.x;
  const int wid  = tid >> 5, lane = tid & 31;
  const int g    = lane >> 4, ln = lane & 15;
  const int wm   = wid & 3,  wn  = wid >> 2;
  const int blockM = blockIdx.y * 128;
  const int blockN = blockIdx.x * 128;

  f32x8 acc[2][4];
#pragma unroll
  for (int ms = 0; ms < 2; ++ms)
#pragma unroll
    for (int ns = 0; ns < 4; ++ns) acc[ms][ns] = (f32x8){};

  const int ar = tid >> 1, ac = (tid & 1) * 16;   // A tile: 128 rows x 32 cols
  const int br = tid >> 3, bc = (tid & 7) * 16;   // B tile: 32 rows x 128 cols

  // async-stage one 128x32 A tile + 32x128 B tile into buffer `buf`
  auto stage = [&](int buf, int k0) {
    const ushort_t* ga = A + (size_t)(blockM + ar) * K + k0 + ac;
    ushort_t* la = &As[buf][ar * APITCH + ac];
    async_ld16(la,      ga);
    async_ld16(la + 8,  ga + 8);
    const ushort_t* gb = Bw + (size_t)(k0 + br) * N + blockN + bc;
    ushort_t* lb = &Bs[buf][br * BPITCH + bc];
    async_ld16(lb,      gb);
    async_ld16(lb + 8,  gb + 8);
  };

  const int nk = K >> 5;
  stage(0, 0);

  for (int kt = 0; kt < nk; ++kt) {
    const int cur = kt & 1;
    wait_async0();        // this wave's stage of buf `cur` has landed in LDS
    __syncthreads();      // everyone's has; also fences last iter's LDS reads
    if (kt + 1 < nk) stage(cur ^ 1, (kt + 1) * 32);

    bf16x16 af[2], bfr[4];
#pragma unroll
    for (int ms = 0; ms < 2; ++ms) {
      const ushort_t* ap = &As[cur][(wm * 32 + ms * 16 + ln) * APITCH];
      af[ms] = make_frag(*(const u32x4*)(ap + 8 * g),
                         *(const u32x4*)(ap + 16 + 8 * g));
    }
#pragma unroll
    for (int ns = 0; ns < 4; ++ns) {
      const ushort_t* bp = &Bs[cur][wn * 64 + ns * 16 + ln];
#pragma unroll
      for (int e = 0; e < 16; ++e)
        bfr[ns][e] = __builtin_bit_cast(__bf16, bp[((e & 7) + 8 * g + 16 * (e >> 3)) * BPITCH]);
    }
#pragma unroll
    for (int ms = 0; ms < 2; ++ms)
#pragma unroll
      for (int ns = 0; ns < 4; ++ns)
        acc[ms][ns] = __builtin_amdgcn_wmma_f32_16x16x32_bf16(
            false, af[ms], false, bfr[ns], (short)0, acc[ms][ns], false, false);
  }

  const float alpha = alpha_p ? alpha_p[0] : 1.0f;
  const float beta  = beta_p  ? beta_p[0]  : 1.0f;
#pragma unroll
  for (int ms = 0; ms < 2; ++ms) {
#pragma unroll
    for (int ns = 0; ns < 4; ++ns) {
#pragma unroll
      for (int i = 0; i < 8; ++i) {
        const int row = blockM + wm * 32 + ms * 16 + i + 8 * g;
        const int col = blockN + wn * 64 + ns * 16 + ln;
        float v = acc[ms][ns][i];
        if (bias) v += bias[col];
        v *= alpha;
        const size_t idx = (size_t)row * N + col;
        if (resid) v += beta * resid[idx];
        if (outf)  outf[idx] = v;
        if (outb)  outb[idx] = f2bf(v);
      }
    }
  }
}

// ---------------------------------------------------------------------------
// Spike cross-attention core.  One block per (b,t,h); 8 waves; wave owns 32 q rows.
// scores = (Q K^T)/8 * mw + ts ; sp = sigmoid(5*(s-max)) ; out = (sp@V)/sum(sp)
// ---------------------------------------------------------------------------
#define KTP 264   // Kt row pitch (256 + 8), 528B, bank-spread
#define VSP 72    // V  row pitch (64 + 8), 144B
#define SPP 264   // sp row pitch
#define ATTN_SMEM ((64*KTP + 256*VSP + 8*16*SPP) * 2)

__global__ __launch_bounds__(256) void attn_kernel(
    const ushort_t* __restrict__ Qb, const ushort_t* __restrict__ Kb,
    const ushort_t* __restrict__ Vb, ushort_t* __restrict__ Ob,
    const float* __restrict__ mw, const float* __restrict__ ts,
    int qi, int ki)
{
  extern __shared__ char smem[];
  const int h = blockIdx.x, t = blockIdx.y, b = blockIdx.z;
  const int tid = threadIdx.x;
  const int wid = tid >> 5, lane = tid & 31;
  const int g = lane >> 4, ln = lane & 15;

  ushort_t* Kt = (ushort_t*)smem;                         // [64][KTP]  K transposed
  ushort_t* Vs = (ushort_t*)(smem + 64 * KTP * 2);        // [256][VSP] V row-major
  ushort_t* Sp = (ushort_t*)(smem + 64 * KTP * 2 + 256 * VSP * 2) + wid * 16 * SPP;

  const float scale = mw[qi * HH + h] * mw[ki * HH + h] * 0.125f;
  const float tsv   = ts[t * HH + h];
  const size_t bt   = ((size_t)b * TT + t) * SS;

  // stage V via async copies and K (transposed) via ds stores; one row/thread
  {
    const int s = tid;
    const ushort_t* krow = Kb + (bt + s) * DD + h * HDIM;
    const ushort_t* vrow = Vb + (bt + s) * DD + h * HDIM;
    ushort_t* vdst = Vs + (size_t)s * VSP;
#pragma unroll
    for (int j = 0; j < 8; ++j)
      async_ld16(vdst + j * 8, vrow + j * 8);
#pragma unroll
    for (int j = 0; j < 8; ++j) {
      u32x4 kq = ((const u32x4*)krow)[j];
      const ushort_t* ke = (const ushort_t*)&kq;
#pragma unroll
      for (int e2 = 0; e2 < 8; ++e2) Kt[(j * 8 + e2) * KTP + s] = ke[e2];
    }
  }
  wait_async0();
  __syncthreads();

  for (int qq = 0; qq < 2; ++qq) {
    const int qt = wid * 2 + qq;               // q tile (16 rows)
    const ushort_t* qrow = Qb + (bt + qt * 16 + ln) * DD + h * HDIM;
    bf16x16 aq0 = make_frag(*(const u32x4*)(qrow + 8 * g),
                            *(const u32x4*)(qrow + 16 + 8 * g));
    bf16x16 aq1 = make_frag(*(const u32x4*)(qrow + 32 + 8 * g),
                            *(const u32x4*)(qrow + 48 + 8 * g));

    // ---- scores: 16 key tiles held in registers -------------------------
    f32x8 acc[16];
#pragma unroll
    for (int kt = 0; kt < 16; ++kt) {
      const ushort_t* p0 = Kt + kt * 16 + ln;                        // d = 0..31
      const ushort_t* p1 = Kt + (size_t)32 * KTP + kt * 16 + ln;     // d = 32..63
      bf16x16 bk0, bk1;
#pragma unroll
      for (int e = 0; e < 16; ++e) {
        const int kd = (e & 7) + 8 * g + 16 * (e >> 3);
        bk0[e] = __builtin_bit_cast(__bf16, p0[(size_t)kd * KTP]);
        bk1[e] = __builtin_bit_cast(__bf16, p1[(size_t)kd * KTP]);
      }
      f32x8 a = (f32x8){};
      a = __builtin_amdgcn_wmma_f32_16x16x32_bf16(false, aq0, false, bk0, (short)0, a, false, false);
      a = __builtin_amdgcn_wmma_f32_16x16x32_bf16(false, aq1, false, bk1, (short)0, a, false, false);
      acc[kt] = a;
    }

    // ---- row max over all 256 keys --------------------------------------
    float rmax[8];
#pragma unroll
    for (int i = 0; i < 8; ++i) rmax[i] = -3.0e38f;
#pragma unroll
    for (int kt = 0; kt < 16; ++kt)
#pragma unroll
      for (int i = 0; i < 8; ++i)
        rmax[i] = fmaxf(rmax[i], acc[kt][i] * scale + tsv);
#pragma unroll
    for (int i = 0; i < 8; ++i)
#pragma unroll
      for (int off = 8; off >= 1; off >>= 1)
        rmax[i] = fmaxf(rmax[i], __shfl_xor(rmax[i], off, 32));

    // ---- sigmoid probs -> LDS (A layout rows), row sums ------------------
    float rsum[8];
#pragma unroll
    for (int i = 0; i < 8; ++i) rsum[i] = 0.0f;
#pragma unroll
    for (int kt = 0; kt < 16; ++kt) {
#pragma unroll
      for (int i = 0; i < 8; ++i) {
        float v = acc[kt][i] * scale + tsv;
        float p = 1.0f / (1.0f + __expf(-5.0f * (v - rmax[i])));
        rsum[i] += p;
        Sp[(i + 8 * g) * SPP + kt * 16 + ln] = f2bf(p);
      }
    }
#pragma unroll
    for (int i = 0; i < 8; ++i) {
#pragma unroll
      for (int off = 8; off >= 1; off >>= 1)
        rsum[i] += __shfl_xor(rsum[i], off, 32);
      rsum[i] = 1.0f / (rsum[i] + 1e-8f);
    }

    // ---- out = sp @ V ----------------------------------------------------
    f32x8 oacc[4];
#pragma unroll
    for (int ns = 0; ns < 4; ++ns) oacc[ns] = (f32x8){};
#pragma unroll
    for (int kk = 0; kk < 8; ++kk) {
      const ushort_t* sr = Sp + (size_t)ln * SPP + kk * 32;
      bf16x16 asp = make_frag(*(const u32x4*)(sr + 8 * g),
                              *(const u32x4*)(sr + 16 + 8 * g));
#pragma unroll
      for (int ns = 0; ns < 4; ++ns) {
        const ushort_t* vp = Vs + (size_t)(kk * 32) * VSP + ns * 16 + ln;
        bf16x16 bv;
#pragma unroll
        for (int e = 0; e < 16; ++e)
          bv[e] = __builtin_bit_cast(__bf16, vp[(size_t)((e & 7) + 8 * g + 16 * (e >> 3)) * VSP]);
        oacc[ns] = __builtin_amdgcn_wmma_f32_16x16x32_bf16(
            false, asp, false, bv, (short)0, oacc[ns], false, false);
      }
    }

    // ---- normalize + store bf16 [B,T,S,D] -------------------------------
#pragma unroll
    for (int ns = 0; ns < 4; ++ns)
#pragma unroll
      for (int i = 0; i < 8; ++i) {
        const int srow = qt * 16 + i + 8 * g;
        float val = oacc[ns][i] * rsum[i];
        Ob[(bt + srow) * DD + h * HDIM + ns * 16 + ln] = f2bf(val);
      }
  }
}

// ---------------------------------------------------------------------------
// activity[b,m,t] = sum over (s,d) of mods[m][b,t]
// ---------------------------------------------------------------------------
__global__ __launch_bounds__(256) void activity_kernel(
    const float* __restrict__ m0, const float* __restrict__ m1, float* __restrict__ act)
{
  const int bx = blockIdx.x;                 // b*64 + m*32 + t
  const int b = bx >> 6, m = (bx >> 5) & 1, t = bx & 31;
  const float* src = (m ? m1 : m0) + ((size_t)(b * TT + t)) * (SS * DD);
  const f32x4* s4 = (const f32x4*)src;
  float s = 0.0f;
  for (int i = threadIdx.x; i < (SS * DD) / 4; i += 256) {
    f32x4 v = s4[i];
    s += v[0] + v[1] + v[2] + v[3];
  }
  __shared__ float red[256];
  red[threadIdx.x] = s;
  __syncthreads();
  for (int o = 128; o >= 1; o >>= 1) {
    if (threadIdx.x < o) red[threadIdx.x] += red[threadIdx.x + o];
    __syncthreads();
  }
  if (threadIdx.x == 0) act[bx] = red[0];
}

// ---------------------------------------------------------------------------
// alignment MLP + softmax -> temporal weights tw[B,NM,T]  (one block, 128 thr)
// ---------------------------------------------------------------------------
__global__ __launch_bounds__(128) void align_kernel(
    const float* __restrict__ act, const float* __restrict__ W1, const float* __restrict__ b1,
    const float* __restrict__ W2, const float* __restrict__ b2,
    const float* __restrict__ toff, float* __restrict__ tw)
{
  __shared__ float flat[2 * 64];
  __shared__ float hsh[2 * 128];
  __shared__ float tot[2 * 64];
  const int tid = threadIdx.x;
  flat[tid] = act[tid];
  __syncthreads();
  for (int b = 0; b < 2; ++b) {
    float a = b1[tid];
    for (int k = 0; k < 64; ++k) a += flat[b * 64 + k] * W1[k * 128 + tid];
    hsh[b * 128 + tid] = fmaxf(a, 0.0f);
  }
  __syncthreads();
  if (tid < 64) {
    for (int b = 0; b < 2; ++b) {
      float a = b2[tid];
      for (int j = 0; j < 128; ++j) a += hsh[b * 128 + j] * W2[j * 64 + tid];
      tot[b * 64 + tid] = toff[tid] + 0.1f * tanhf(a);
    }
  }
  __syncthreads();
  // softmax over T=32 per (b,m): tid = b*64 + m*32 + t, each group is one wave32
  float v = tot[tid];
  float m = v;
  for (int off = 16; off >= 1; off >>= 1) m = fmaxf(m, __shfl_xor(m, off, 32));
  float e = __expf(v - m);
  float s = e;
  for (int off = 16; off >= 1; off >>= 1) s += __shfl_xor(s, off, 32);
  tw[tid] = e / s;
}

// ---------------------------------------------------------------------------
// scale modality by temporal weight; produce f32 (in place) + bf16 copies
// ---------------------------------------------------------------------------
__global__ void scale_kernel(float* __restrict__ mf, ushort_t* __restrict__ mb,
                             const float* __restrict__ tw, int mod, int n)
{
  int i = blockIdx.x * blockDim.x + threadIdx.x;
  const int stride = gridDim.x * blockDim.x;
  for (; i < n; i += stride) {
    const int b = i >> 22;            // T*S*D = 2^22
    const int t = (i >> 17) & 31;     // S*D   = 2^17
    const float w = tw[b * 64 + mod * 32 + t];
    const float v = mf[i] * w;
    mf[i] = v;
    mb[i] = f2bf(v);
  }
}

// ---------------------------------------------------------------------------
// Host orchestration
// ---------------------------------------------------------------------------
extern "C" void kernel_launch(void* const* d_in, const int* in_sizes, int n_in,
                              void* d_out, int out_size, void* d_ws, size_t ws_size,
                              hipStream_t stream) {
  (void)in_sizes; (void)n_in; (void)out_size; (void)ws_size;

  // ---- input pointers (insertion order of setup_inputs / params dicts) ----
  const float* xv   = (const float*)d_in[0];
  const float* xa   = (const float*)d_in[1];
  const float* pW[2]  = { (const float*)d_in[2], (const float*)d_in[4] };
  const float* pb[2]  = { (const float*)d_in[3], (const float*)d_in[5] };
  const float* toff = (const float*)d_in[6];
  const float* aW1  = (const float*)d_in[7];
  const float* ab1  = (const float*)d_in[8];
  const float* aW2  = (const float*)d_in[9];
  const float* ab2  = (const float*)d_in[10];
  // xattn layer p base = 11 + p*10 : Wq,bq,Wk,bk,Wv,bv,Wo,bo,mw,ts
  const float* Wx[2][4]; const float* bx[2][4];
  const float* mwp[2]; const float* tsp[2];
  for (int p = 0; p < 2; ++p) {
    const int base = 11 + p * 10;
    for (int w = 0; w < 4; ++w) {
      Wx[p][w] = (const float*)d_in[base + 2 * w];
      bx[p][w] = (const float*)d_in[base + 2 * w + 1];
    }
    mwp[p] = (const float*)d_in[base + 8];
    tsp[p] = (const float*)d_in[base + 9];
  }
  const float* fw = (const float*)d_in[31];

  // ---- workspace carving ----
  char* ws = (char*)d_ws;
  const size_t WSLOT = (size_t)DD * DD * 2;   // one bf16 weight matrix
  const size_t BF_T  = NTEL * 2;              // bf16 tensor
  const size_t F_T   = NTEL * 4;              // f32 tensor
  ushort_t* Wb[10];
  for (int i = 0; i < 10; ++i) Wb[i] = (ushort_t*)(ws + i * WSLOT);
  size_t off = 10 * WSLOT;
  ushort_t* xb  = (ushort_t*)(ws + off); off += BF_T;
  ushort_t* yb  = (ushort_t*)(ws + off); off += BF_T;
  ushort_t* mb0 = (ushort_t*)(ws + off); off += BF_T;
  ushort_t* mb1 = (ushort_t*)(ws + off); off += BF_T;
  ushort_t* Qb  = (ushort_t*)(ws + off); off += BF_T;
  ushort_t* Kb  = (ushort_t*)(ws + off); off += BF_T;
  ushort_t* Vb  = (ushort_t*)(ws + off); off += BF_T;
  ushort_t* Ab  = (ushort_t*)(ws + off); off += BF_T;
  float* mf0 = (float*)(ws + off); off += F_T;
  float* mf1 = (float*)(ws + off); off += F_T;
  float* act = (float*)(ws + off); off += 1024;
  float* twb = (float*)(ws + off); off += 1024;

  ushort_t* mb[2] = { mb0, mb1 };
  float*    mf[2] = { mf0, mf1 };

  const dim3 gemm_grid(DD / 128, NTOK / 128);     // (4, 128)
  const dim3 attn_grid(HH, TT, BB);
  const int NE = (int)NTEL;
  const int NW = DD * DD;

  auto GEMM = [&](const ushort_t* Am, const ushort_t* Bm, const float* bias,
                  const float* resid, const float* ap, const float* bp,
                  float* of, ushort_t* ob) {
    gemm_bf16<<<gemm_grid, 256, 0, stream>>>(Am, Bm, bias, resid, ap, bp, of, ob,
                                             NTOK, DD, DD);
  };
  auto CVT = [&](const float* in, ushort_t* out, int n, int blocks) {
    cvt_f32_bf16<<<blocks, 256, 0, stream>>>(in, out, n);
  };

  // 1) convert inputs + weights to bf16
  CVT(xv, xb, NE, 4096);
  CVT(xa, yb, NE, 4096);
  CVT(pW[0], Wb[0], NW, 256);
  CVT(pW[1], Wb[1], NW, 256);
  for (int p = 0; p < 2; ++p)
    for (int w = 0; w < 4; ++w)
      CVT(Wx[p][w], Wb[2 + p * 4 + w], NW, 256);

  // 2) modality projections -> f32 mods
  GEMM(xb, Wb[0], pb[0], nullptr, nullptr, nullptr, mf0, nullptr);
  GEMM(yb, Wb[1], pb[1], nullptr, nullptr, nullptr, mf1, nullptr);

  // 3) adaptive temporal alignment
  activity_kernel<<<BB * 2 * TT, 256, 0, stream>>>(mf0, mf1, act);
  align_kernel<<<1, 128, 0, stream>>>(act, aW1, ab1, aW2, ab2, toff, twb);
  scale_kernel<<<4096, 256, 0, stream>>>(mf0, mb0, twb, 0, NE);
  scale_kernel<<<4096, 256, 0, stream>>>(mf1, mb1, twb, 1, NE);

  // 4) cross-modal enhancement: e_i (in mf[i]) += xattn_p(e_i, mods_j) for p=0,1
  for (int i = 0; i < 2; ++i) {
    const int j = 1 - i;
    for (int p = 0; p < 2; ++p) {
      CVT(mf[i], xb, NE, 4096);                                   // e_i -> bf16
      GEMM(xb,    Wb[2 + p * 4 + 0], bx[p][0], nullptr, nullptr, nullptr, nullptr, Qb);
      GEMM(mb[j], Wb[2 + p * 4 + 1], bx[p][1], nullptr, nullptr, nullptr, nullptr, Kb);
      GEMM(mb[j], Wb[2 + p * 4 + 2], bx[p][2], nullptr, nullptr, nullptr, nullptr, Vb);
      attn_kernel<<<attn_grid, 256, ATTN_SMEM, stream>>>(Qb, Kb, Vb, Ab, mwp[p], tsp[p], i, j);
      GEMM(Ab, Wb[2 + p * 4 + 3], bx[p][3], mf[i], nullptr, nullptr, mf[i], nullptr);
    }
  }

  // 5) final fusion: att = xattn_0(e0, e1); out = fw0*e0 + fw1*att
  CVT(mf0, xb, NE, 4096);
  CVT(mf1, yb, NE, 4096);
  GEMM(xb, Wb[2], bx[0][0], nullptr, nullptr, nullptr, nullptr, Qb);
  GEMM(yb, Wb[3], bx[0][1], nullptr, nullptr, nullptr, nullptr, Kb);
  GEMM(yb, Wb[4], bx[0][2], nullptr, nullptr, nullptr, nullptr, Vb);
  attn_kernel<<<attn_grid, 256, ATTN_SMEM, stream>>>(Qb, Kb, Vb, Ab, mwp[0], tsp[0], 0, 1);
  GEMM(Ab, Wb[5], bx[0][3], mf0, fw + 1, fw + 0, (float*)d_out, nullptr);
}
